// MultiheadSelfAttention_6399501271478
// MI455X (gfx1250) — matrix operand order repaired and checked by reading, not compiled, not run on hardware
//
#include <hip/hip_runtime.h>

// ---------------------------------------------------------------------------
// MultiheadSelfAttention with RoPE for MI455X (gfx1250), wave32 WMMA bf16.
// B=2, S=2048, D_MODEL=1024, H=16, D_K=64.
// Round 3: cross-iteration software pipelining (ping-pong fragment buffers
// with compile-time indices) + waves_per_eu hint so WMMAs overlap L2 loads.
// ---------------------------------------------------------------------------

typedef __attribute__((ext_vector_type(16))) __bf16 v16bf;
typedef __attribute__((ext_vector_type(8)))  __bf16 v8bf;
typedef __attribute__((ext_vector_type(8)))  float  v8f;

#define WMMA_BF16(a, b, c) \
  __builtin_amdgcn_wmma_f32_16x16x32_bf16(false, (a), false, (b), (short)0, (c), false, false)

// ---- fragment loaders -------------------------------------------------------
// A-matrix 16x32 bf16, row-major source (lda elements):
//   lane<16 : row = row0+l16, K = {0..7, 16..23}
//   lane>=16: row = row0+l16, K = {8..15, 24..31}
static __device__ inline v16bf load_a_rowmajor(const __bf16* base, int lda,
                                               int row0, int k0, int l16, int lh) {
  const __bf16* p = base + (size_t)(row0 + l16) * lda + k0 + lh * 8;
  union { v16bf v; v8bf h[2]; } u;
  u.h[0] = *(const v8bf*)(p);
  u.h[1] = *(const v8bf*)(p + 16);
  return u.v;
}

// B-matrix 32x16 bf16 where column n is contiguous in memory at colbase + n*ldb:
//   lane<16 : col = n0+l16, K = {k0..k0+7, k0+16..k0+23}
//   lane>=16: col = n0+l16, K = {k0+8..k0+15, k0+24..k0+31}
static __device__ inline v16bf load_b_coldata(const __bf16* colbase, int ldb,
                                              int n0, int k0, int l16, int lh) {
  const __bf16* p = colbase + (size_t)(n0 + l16) * ldb + k0 + lh * 8;
  union { v16bf v; v8bf h[2]; } u;
  u.h[0] = *(const v8bf*)(p);
  u.h[1] = *(const v8bf*)(p + 16);
  return u.v;
}

// xor-shuffle reductions within each 16-lane group (wave32)
static __device__ inline float redmax16(float v) {
  v = fmaxf(v, __shfl_xor(v, 1, 32));
  v = fmaxf(v, __shfl_xor(v, 2, 32));
  v = fmaxf(v, __shfl_xor(v, 4, 32));
  v = fmaxf(v, __shfl_xor(v, 8, 32));
  return v;
}
static __device__ inline float redsum16(float v) {
  v += __shfl_xor(v, 1, 32);
  v += __shfl_xor(v, 2, 32);
  v += __shfl_xor(v, 4, 32);
  v += __shfl_xor(v, 8, 32);
  return v;
}

// ---- 0) f32 -> bf16 conversion ---------------------------------------------
__global__ void mhsa_cvt_bf16(const float* __restrict__ src,
                              __bf16* __restrict__ dst, int n) {
  int i = blockIdx.x * 256 + threadIdx.x;
  if (i < n) dst[i] = (__bf16)src[i];
}

// ---- 1) fused QKV projection + RoPE ----------------------------------------
// One pipeline stage: consume stage-k fragments (A in abuf[CUR], Wq group in
// bq[CUR], Wk/Wv groups loaded at stage top), prefetch stage-(k+32) A + Wq
// group into the CUR^1 buffers.  All buffer indices are compile-time.
template<int CUR, bool PRE>
static __device__ inline void proj_stage(
    int k0, int n0, int m0, int l16, int lh,
    const __bf16* __restrict__ Xb, const __bf16* __restrict__ Wqb,
    const __bf16* __restrict__ Wkb, const __bf16* __restrict__ Wvb,
    v16bf (&abuf)[2], v16bf (&bq)[2][4],
    v8f (&aq)[4], v8f (&ak)[4], v8f (&av)[4]) {
  v16bf bk[4], bv[4];
#pragma unroll
  for (int t = 0; t < 4; ++t) bk[t] = load_b_coldata(Wkb, 1024, n0 + t * 16, k0, l16, lh);
#pragma unroll
  for (int t = 0; t < 4; ++t) bv[t] = load_b_coldata(Wvb, 1024, n0 + t * 16, k0, l16, lh);
  if (PRE) {
    abuf[CUR ^ 1] = load_a_rowmajor(Xb, 1024, m0, k0 + 32, l16, lh);
#pragma unroll
    for (int t = 0; t < 4; ++t)
      bq[CUR ^ 1][t] = load_b_coldata(Wqb, 1024, n0 + t * 16, k0 + 32, l16, lh);
  }
#pragma unroll
  for (int t = 0; t < 4; ++t) aq[t] = WMMA_BF16(abuf[CUR], bq[CUR][t], aq[t]);
#pragma unroll
  for (int t = 0; t < 4; ++t) ak[t] = WMMA_BF16(abuf[CUR], bk[t], ak[t]);
#pragma unroll
  for (int t = 0; t < 4; ++t) av[t] = WMMA_BF16(abuf[CUR], bv[t], av[t]);
}

// grid (32, 16) = (BS/128, heads), block 256 (8 waves = 2 waves/SIMD on a WGP).
__global__ __launch_bounds__(256, 2) void mhsa_proj_qkv(
    const __bf16* __restrict__ Xb,
    const __bf16* __restrict__ Wqb, const __bf16* __restrict__ Wkb,
    const __bf16* __restrict__ Wvb,
    __bf16* __restrict__ Qo, __bf16* __restrict__ Ko, __bf16* __restrict__ Vto) {
  const int lane = threadIdx.x & 31, wave = threadIdx.x >> 5;
  const int l16 = lane & 15, lh = (lane >> 4) & 1;
  const int h  = blockIdx.y;
  const int n0 = h * 64;
  const int m0 = blockIdx.x * 128 + wave * 16;

  v8f aq[4] = {}, ak[4] = {}, av[4] = {};
  v16bf abuf[2];
  v16bf bq[2][4];

  // pipeline prologue: stage-0 A and Wq group
  abuf[0] = load_a_rowmajor(Xb, 1024, m0, 0, l16, lh);
#pragma unroll
  for (int t = 0; t < 4; ++t) bq[0][t] = load_b_coldata(Wqb, 1024, n0 + t * 16, 0, l16, lh);

  for (int k0 = 0; k0 < 960; k0 += 64) {
    proj_stage<0, true>(k0,      n0, m0, l16, lh, Xb, Wqb, Wkb, Wvb, abuf, bq, aq, ak, av);
    proj_stage<1, true>(k0 + 32, n0, m0, l16, lh, Xb, Wqb, Wkb, Wvb, abuf, bq, aq, ak, av);
  }
  proj_stage<0, true >(960, n0, m0, l16, lh, Xb, Wqb, Wkb, Wvb, abuf, bq, aq, ak, av);
  proj_stage<1, false>(992, n0, m0, l16, lh, Xb, Wqb, Wkb, Wvb, abuf, bq, aq, ak, av);

  // Epilogue: RoPE on Q,K (even/odd head-dim pairs live in adjacent lanes),
  // then store Q,K row-major [bh][s][d] and V transposed [bh][d][s], all bf16.
#pragma unroll
  for (int t = 0; t < 4; ++t) {
    const int dcol = t * 16 + l16;                 // head-local dim 0..63
    const float invf = __expf(-(float)(dcol & ~1) * (9.21034037198f / 64.0f));
    const bool even = (dcol & 1) == 0;
#pragma unroll
    for (int r = 0; r < 8; ++r) {
      const int m = m0 + r + 8 * lh;               // global row over B*S
      const int s = m & 2047, b = m >> 11;
      const int bh = b * 16 + h;
      float sn, cs;
      __sincosf((float)s * invf, &sn, &cs);
      float q = aq[t][r], qp = __shfl_xor(q, 1, 32);
      float k = ak[t][r], kp = __shfl_xor(k, 1, 32);
      float qn = even ? (q * cs - qp * sn) : (qp * sn + q * cs);
      float kn = even ? (k * cs - kp * sn) : (kp * sn + k * cs);
      Qo[(size_t)(bh * 2048 + s) * 64 + dcol] = (__bf16)qn;
      Ko[(size_t)(bh * 2048 + s) * 64 + dcol] = (__bf16)kn;
      Vto[(size_t)(bh * 64 + dcol) * 2048 + s] = (__bf16)av[t][r];
    }
  }
}

// ---- 2) causal flash attention ---------------------------------------------
// Process one 32-key chunk resident in (kc, vc); prefetch the next chunk into
// (kn, vn) first so its L2 latency hides behind 8 WMMAs + softmax VALU work.
static __device__ inline void attn_chunk(
    int kb, int q0, int l16, int lh,
    const __bf16* __restrict__ Kp, const __bf16* __restrict__ Vp, __bf16* Lp,
    const v16bf& aQ0, const v16bf& aQ1,
    v16bf (&kc)[4], v16bf (&vc)[4],
    v16bf (&kn)[4], v16bf (&vn)[4], bool pre,
    v8f (&o)[4], float (&mr)[8], float (&lr)[8]) {
  if (pre) {
    kn[0] = load_b_coldata(Kp, 64, kb + 32, 0,  l16, lh);
    kn[1] = load_b_coldata(Kp, 64, kb + 32, 32, l16, lh);
    kn[2] = load_b_coldata(Kp, 64, kb + 48, 0,  l16, lh);
    kn[3] = load_b_coldata(Kp, 64, kb + 48, 32, l16, lh);
#pragma unroll
    for (int t = 0; t < 4; ++t)
      vn[t] = load_b_coldata(Vp, 2048, t * 16, kb + 32, l16, lh);
  }

  v8f s0 = {}, s1 = {};
  s0 = WMMA_BF16(aQ0, kc[0], s0);
  s0 = WMMA_BF16(aQ1, kc[1], s0);
  s1 = WMMA_BF16(aQ0, kc[2], s1);
  s1 = WMMA_BF16(aQ1, kc[3], s1);

  const int key0 = kb + l16, key1 = kb + 16 + l16;
#pragma unroll
  for (int r = 0; r < 8; ++r) {
    const int q = q0 + r + 8 * lh;
    float v0 = s0[r] * 0.125f; if (key0 > q) v0 = -1.0e30f;   // 1/sqrt(64)
    float v1 = s1[r] * 0.125f; if (key1 > q) v1 = -1.0e30f;
    const float rm = redmax16(fmaxf(v0, v1));
    const float mn = fmaxf(mr[r], rm);
    const float corr = __expf(mr[r] - mn);
    const float p0 = __expf(v0 - mn);
    const float p1 = __expf(v1 - mn);
    const float rs = redsum16(p0 + p1);
    lr[r] = lr[r] * corr + rs;
    mr[r] = mn;
#pragma unroll
    for (int t = 0; t < 4; ++t) o[t][r] *= corr;
    const int mrow = r + 8 * lh;                    // C-layout -> LDS (m, k)
    Lp[mrow * 32 + l16]      = (__bf16)p0;
    Lp[mrow * 32 + 16 + l16] = (__bf16)p1;
  }

  // re-read P in A-fragment layout (intra-wave LDS: in-order, no barrier)
  union { v16bf v; v8bf hh[2]; } u;
  const __bf16* pp = Lp + l16 * 32 + lh * 8;
  u.hh[0] = *(const v8bf*)(pp);
  u.hh[1] = *(const v8bf*)(pp + 16);

#pragma unroll
  for (int t = 0; t < 4; ++t)                       // O += P (16x32) * V (32x16)
    o[t] = WMMA_BF16(u.v, vc[t], o[t]);
}

// grid (16, 32) = (S/128, B*H), block 256 (8 waves, 16 query rows each).
__global__ __launch_bounds__(256, 2) void mhsa_attn(
    const __bf16* __restrict__ Qb, const __bf16* __restrict__ Kb,
    const __bf16* __restrict__ Vtb, __bf16* __restrict__ Hd) {
  __shared__ __bf16 plds[8][16 * 32];   // per-wave 16x32 P-tile transpose staging
  const int lane = threadIdx.x & 31, wave = threadIdx.x >> 5;
  const int l16 = lane & 15, lh = (lane >> 4) & 1;
  const int bh = blockIdx.y;
  const int q0 = blockIdx.x * 128 + wave * 16;

  const __bf16* Qp = Qb + (size_t)bh * 2048 * 64;
  const __bf16* Kp = Kb + (size_t)bh * 2048 * 64;
  const __bf16* Vp = Vtb + (size_t)bh * 64 * 2048;
  __bf16* Lp = &plds[wave][0];

  const v16bf aQ0 = load_a_rowmajor(Qp, 64, q0, 0, l16, lh);
  const v16bf aQ1 = load_a_rowmajor(Qp, 64, q0, 32, l16, lh);

  v8f o[4] = {};
  float mr[8], lr[8];
#pragma unroll
  for (int r = 0; r < 8; ++r) { mr[r] = -3.0e38f; lr[r] = 0.0f; }

  // pipeline prologue: chunk 0 fragments
  v16bf kA[4], vA[4], kB[4], vB[4];
  kA[0] = load_b_coldata(Kp, 64, 0,  0,  l16, lh);
  kA[1] = load_b_coldata(Kp, 64, 0,  32, l16, lh);
  kA[2] = load_b_coldata(Kp, 64, 16, 0,  l16, lh);
  kA[3] = load_b_coldata(Kp, 64, 16, 32, l16, lh);
#pragma unroll
  for (int t = 0; t < 4; ++t) vA[t] = load_b_coldata(Vp, 2048, t * 16, 0, l16, lh);

  const int kmax = q0 + 15;                         // causal bound (inclusive)
  int kb = 0;
  while (true) {
    bool pre = (kb + 32) <= kmax;
    attn_chunk(kb, q0, l16, lh, Kp, Vp, Lp, aQ0, aQ1, kA, vA, kB, vB, pre, o, mr, lr);
    if (!pre) break;
    kb += 32;
    pre = (kb + 32) <= kmax;
    attn_chunk(kb, q0, l16, lh, Kp, Vp, Lp, aQ0, aQ1, kB, vB, kA, vA, pre, o, mr, lr);
    if (!pre) break;
    kb += 32;
  }

  const int b = bh >> 4, h = bh & 15;
#pragma unroll
  for (int t = 0; t < 4; ++t) {
#pragma unroll
    for (int r = 0; r < 8; ++r) {
      const int q = q0 + r + 8 * lh;
      const float val = o[t][r] / lr[r];
      Hd[(size_t)(b * 2048 + q) * 1024 + h * 64 + t * 16 + l16] = (__bf16)val;
    }
  }
}

// ---- 3) output projection: out = head * W_o^T  (f32 result) ----------------
template<int CUR, bool PRE>
static __device__ inline void outp_stage(
    int k0, int nb, int m0, int l16, int lh,
    const __bf16* __restrict__ Hd, const __bf16* __restrict__ Wob,
    v16bf (&abuf)[2], v16bf (&bbuf)[2][4], v8f (&acc)[4]) {
  if (PRE) {
    abuf[CUR ^ 1] = load_a_rowmajor(Hd, 1024, m0, k0 + 32, l16, lh);
#pragma unroll
    for (int t = 0; t < 4; ++t)
      bbuf[CUR ^ 1][t] = load_b_coldata(Wob, 1024, nb + t * 16, k0 + 32, l16, lh);
  }
#pragma unroll
  for (int t = 0; t < 4; ++t) acc[t] = WMMA_BF16(abuf[CUR], bbuf[CUR][t], acc[t]);
}

// grid (32, 16), block 256; wave computes 16 rows x 64 cols.
__global__ __launch_bounds__(256, 2) void mhsa_out_proj(
    const __bf16* __restrict__ Hd, const __bf16* __restrict__ Wob,
    float* __restrict__ Out) {
  const int lane = threadIdx.x & 31, wave = threadIdx.x >> 5;
  const int l16 = lane & 15, lh = (lane >> 4) & 1;
  const int m0 = blockIdx.x * 128 + wave * 16;
  const int nb = blockIdx.y * 64;

  v8f acc[4] = {};
  v16bf abuf[2];
  v16bf bbuf[2][4];
  abuf[0] = load_a_rowmajor(Hd, 1024, m0, 0, l16, lh);
#pragma unroll
  for (int t = 0; t < 4; ++t) bbuf[0][t] = load_b_coldata(Wob, 1024, nb + t * 16, 0, l16, lh);

  for (int k0 = 0; k0 < 960; k0 += 64) {
    outp_stage<0, true>(k0,      nb, m0, l16, lh, Hd, Wob, abuf, bbuf, acc);
    outp_stage<1, true>(k0 + 32, nb, m0, l16, lh, Hd, Wob, abuf, bbuf, acc);
  }
  outp_stage<0, true >(960, nb, m0, l16, lh, Hd, Wob, abuf, bbuf, acc);
  outp_stage<1, false>(992, nb, m0, l16, lh, Hd, Wob, abuf, bbuf, acc);

#pragma unroll
  for (int t = 0; t < 4; ++t) {
#pragma unroll
    for (int r = 0; r < 8; ++r) {
      const int m = m0 + r + 8 * lh;
      Out[(size_t)m * 1024 + nb + t * 16 + l16] = acc[t][r];
    }
  }
}

// ---------------------------------------------------------------------------
extern "C" void kernel_launch(void* const* d_in, const int* in_sizes, int n_in,
                              void* d_out, int out_size, void* d_ws, size_t ws_size,
                              hipStream_t stream) {
  const float* X  = (const float*)d_in[0];   // (2, 2048, 1024)
  const float* Wq = (const float*)d_in[1];   // (1024, 1024) row-major (o, d)
  const float* Wk = (const float*)d_in[2];
  const float* Wv = (const float*)d_in[3];
  const float* Wo = (const float*)d_in[4];
  float* Out = (float*)d_out;

  char* ws = (char*)d_ws;
  // workspace layout (bytes)
  __bf16* Xb  = (__bf16*)(ws + 0);           // 4096*1024*2 =  8,388,608
  __bf16* Wqb = (__bf16*)(ws + 8388608);     // 1024*1024*2 =  2,097,152
  __bf16* Wkb = (__bf16*)(ws + 10485760);
  __bf16* Wvb = (__bf16*)(ws + 12582912);
  __bf16* Wob = (__bf16*)(ws + 14680064);
  __bf16* Qb  = (__bf16*)(ws + 16777216);    // 32*2048*64*2 = 8,388,608
  __bf16* Kb  = (__bf16*)(ws + 25165824);
  __bf16* Vtb = (__bf16*)(ws + 33554432);
  __bf16* Hd  = (__bf16*)(ws + 41943040);    // 4096*1024*2 = 8,388,608

  const int nX = 2 * 2048 * 1024;
  const int nW = 1024 * 1024;
  mhsa_cvt_bf16<<<(nX + 255) / 256, 256, 0, stream>>>(X,  Xb,  nX);
  mhsa_cvt_bf16<<<(nW + 255) / 256, 256, 0, stream>>>(Wq, Wqb, nW);
  mhsa_cvt_bf16<<<(nW + 255) / 256, 256, 0, stream>>>(Wk, Wkb, nW);
  mhsa_cvt_bf16<<<(nW + 255) / 256, 256, 0, stream>>>(Wv, Wvb, nW);
  mhsa_cvt_bf16<<<(nW + 255) / 256, 256, 0, stream>>>(Wo, Wob, nW);

  mhsa_proj_qkv<<<dim3(32, 16), 256, 0, stream>>>(Xb, Wqb, Wkb, Wvb, Qb, Kb, Vtb);
  mhsa_attn    <<<dim3(16, 32), 256, 0, stream>>>(Qb, Kb, Vtb, Hd);
  mhsa_out_proj<<<dim3(32, 16), 256, 0, stream>>>(Hd, Wob, Out);
}
